// EnsembleDRarefSolver_343597384512
// MI455X (gfx1250) — compile-verified, matrix operand order checked by reading, not run on hardware
//
#include <hip/hip_runtime.h>
#include <math.h>

// ---------------------------------------------------------------------------
// Fused relativistic-Riemann + ensemble-MLP kernel for gfx1250 (MI455X).
//  - MLP (6->32->32->1, 5 experts) via v_wmma_f32_16x16x32_f16, one wave per
//    32-cell group (two 16-row M tiles), weights as register B-fragments.
//  - D->A relayout between MLP layers bounced through LDS (ds_load_b128),
//    intra-wave ordering enforced with s_wait_dscnt 0.
//  - All transcendentals on the hardware TRANS pipe (v_log/v_exp/v_rcp/
//    v_rsq/v_sqrt/v_tanh) -- inputs are strictly positive & bounded, so the
//    branchy libm range-reduction is pure overhead.
//  - Fully fused: the (NC x E) xi intermediate lives in LDS, never in HBM.
// ---------------------------------------------------------------------------

typedef __attribute__((ext_vector_type(16))) _Float16 v16h;
typedef __attribute__((ext_vector_type(8)))  float    v8f;

#define GAMMA_F 1.6666666666666667f
#define SG_F    0.81649658092772603f   // sqrt(GAMMA-1)
#define LN2_F   0.69314718055994531f
#define LOG2E_F 1.44269504088896341f

union AU { v16h v; _Float16 h[16]; };

// ---- native transcendental helpers (single TRANS32 instruction each) ------
__device__ __forceinline__ float nlog2(float x) { return __builtin_amdgcn_logf(x); }
__device__ __forceinline__ float nexp2(float x) { return __builtin_amdgcn_exp2f(x); }
__device__ __forceinline__ float nrcp (float x) { return __builtin_amdgcn_rcpf(x); }
__device__ __forceinline__ float nrsq (float x) { return __builtin_amdgcn_rsqf(x); }
__device__ __forceinline__ float nsqrt(float x) { return __builtin_amdgcn_sqrtf(x); }
__device__ __forceinline__ float nln  (float x) { return LN2_F * nlog2(x); }
__device__ __forceinline__ float fpow (float x, float y) { return nexp2(y * nlog2(x)); }
__device__ __forceinline__ float nsigmoid(float x) {
  return nrcp(1.f + nexp2(-x * LOG2E_F));
}
__device__ __forceinline__ float ntanh(float x) {
#if __has_builtin(__builtin_amdgcn_tanhf)
  return __builtin_amdgcn_tanhf(x);      // gfx1250 v_tanh_f32 (TRANS32)
#else
  float t = nexp2(2.f * LOG2E_F * x);
  return (t - 1.f) * nrcp(t + 1.f);
#endif
}

__device__ __forceinline__ float csnd(float rho, float p) {
  float h = 1.f + GAMMA_F / (GAMMA_F - 1.f) * p / rho;
  return nsqrt(GAMMA_F * p * nrcp(rho * h));
}
__device__ __forceinline__ float vstar_from_cs(float cs, float cs_a, float v_a,
                                               float wave_sign) {
  float ratio = (SG_F - cs) * nrcp(SG_F + cs) * ((SG_F + cs_a) * nrcp(SG_F - cs_a));
  float A = (1.f + v_a) * nrcp(1.f - v_a) * fpow(ratio, -wave_sign * 2.f / SG_F);
  return (A - 1.f) * nrcp(A + 1.f);
}

// Bisection with loop-invariants hoisted:
//   f(cs) = C1 * ((SG-cs)/(SG+cs))^pexp * R0^pexp  mapped to
//   A = C1 * exp2(pexp * (L0 + log2((SG-cs)/(SG+cs)))) ; f = (A-1)/(A+1)-sign*cs
__device__ __forceinline__ void raref_solver(float rho_a, float p_a, float v_a,
                                             float sign, float& rho, float& p,
                                             float& h, float& v) {
  const float cs_a = csnd(rho_a, p_a);
  const float ws   = -sign;
  const float C1   = (1.f + v_a) * nrcp(1.f - v_a);
  const float L0   = nlog2((SG_F + cs_a) * nrcp(SG_F - cs_a));
  const float pexp = -ws * 2.f / SG_F;

  auto feval = [&](float cs) -> float {
    float rc = (SG_F - cs) * nrcp(SG_F + cs);
    float A  = C1 * nexp2(pexp * (L0 + nlog2(rc)));
    return (A - 1.f) * nrcp(A + 1.f) - sign * cs;
  };

  float lo = 1e-6f, hib = SG_F - 1e-6f;
  float flo = feval(lo);
  for (int it = 0; it < 40; ++it) {
    float mid = 0.5f * (lo + hib);
    float fm = feval(mid);
    int sm = (fm > 0.f) - (fm < 0.f);
    int sl = (flo > 0.f) - (flo < 0.f);
    bool same = (sm == sl);
    lo  = same ? mid : lo;
    flo = same ? fm : flo;
    hib = same ? hib : mid;
  }
  float cs = 0.5f * (lo + hib);
  h = (GAMMA_F - 1.f) * nrcp(GAMMA_F - 1.f - cs * cs);
  float k = (h - 1.f) * (GAMMA_F - 1.f) / GAMMA_F;
  rho = fpow(k * fpow(rho_a, GAMMA_F) * nrcp(p_a), 1.f / (GAMMA_F - 1.f));
  p = k * rho;
  v = sign * cs;
}

#define E_EXP 5
#define HH 32

__global__ __launch_bounds__(256) void fused_riemann_mlp(
    const float* __restrict__ P, const float* __restrict__ F,
    const float* __restrict__ W1, const float* __restrict__ b1,
    const float* __restrict__ W2, const float* __restrict__ b2,
    const float* __restrict__ W3, const float* __restrict__ b3,
    float* __restrict__ out, int nc, int ngroups) {
  // per-wave LDS staging (8 waves / block)
  __shared__ __align__(16) _Float16 h_buf[8][16][HH];   // 8 KB
  __shared__ float xi_buf[8][32][E_EXP];                // 5 KB

  const int lane = threadIdx.x & 31;
  const int wib  = threadIdx.x >> 5;
  const int g    = (blockIdx.x << 3) + wib;   // one 32-cell group per wave
  if (g >= ngroups) return;

  const int hi = (lane >> 4) & 1;   // half-wave select
  const int mn = lane & 15;         // M row (A) or N col (B/C/D)

  // ---------------- Phase 1: ensemble MLP -> xi (WMMA) -------------------
  // A fragments for the two 16-cell subtiles: features
  // [ln rhoL, ln rhoR, ln pL, ln pR, vL, vR] zero-padded to K=32.
  v16h afrag[2];
#pragma unroll
  for (int t = 0; t < 2; ++t) {
    int c = g * 32 + t * 16 + mn;
    if (c >= nc) c = nc - 1;
    float feat[6];
    feat[0] = nln(P[c * 6 + 0]);
    feat[1] = nln(P[c * 6 + 1]);
    feat[2] = nln(P[c * 6 + 2]);
    feat[3] = nln(P[c * 6 + 3]);
    feat[4] = P[c * 6 + 4];
    feat[5] = P[c * 6 + 5];
    AU a;
#pragma unroll
    for (int i = 0; i < 16; ++i) {
      float v = (hi == 0 && i < 6) ? feat[i] : 0.f;
      a.h[i] = (_Float16)v;
    }
    afrag[t] = a.v;
  }

  for (int e = 0; e < E_EXP; ++e) {
    // B fragments (K x N layout mirror of A; K=(i<8?i:i+8)+8*hi, N=lane&15)
    v16h b1f[2], b2f[2];
#pragma unroll
    for (int nt = 0; nt < 2; ++nt) {
      const int row = mn + 16 * nt;   // hidden unit index
      AU bA, bB;
#pragma unroll
      for (int i = 0; i < 16; ++i) {
        const int K = (i < 8 ? i : i + 8) + 8 * hi;
        bA.h[i] = (K < 6) ? (_Float16)W1[(e * HH + row) * 6 + K] : (_Float16)0.f;
        bB.h[i] = (_Float16)W2[(e * HH + row) * HH + K];
      }
      b1f[nt] = bA.v;
      b2f[nt] = bB.v;
    }

#pragma unroll
    for (int t = 0; t < 2; ++t) {
      // ---- layer 1: D = tanh(A @ W1^T + b1), D layout M=r+8*hi, N=mn ----
      v8f czero = {};
      v8f c0 = __builtin_amdgcn_wmma_f32_16x16x32_f16(
          false, afrag[t], false, b1f[0], (short)0, czero, false, false);
      v8f c1 = __builtin_amdgcn_wmma_f32_16x16x32_f16(
          false, afrag[t], false, b1f[1], (short)0, czero, false, false);
      const float bia0 = b1[e * HH + mn];
      const float bia1 = b1[e * HH + mn + 16];
#pragma unroll
      for (int r = 0; r < 8; ++r) {
        h_buf[wib][r + 8 * hi][mn]      = (_Float16)ntanh(c0[r] + bia0);
        h_buf[wib][r + 8 * hi][mn + 16] = (_Float16)ntanh(c1[r] + bia1);
      }
      asm volatile("s_wait_dscnt 0" ::: "memory");   // cross-lane LDS RAW

      // ---- relayout D->A: two contiguous 16B runs per lane -------------
      AU a2;
      const _Float16* hrow = &h_buf[wib][mn][0];
      *(int4*)&a2.h[0] = *(const int4*)&hrow[8 * hi];        // K= 8*hi..+7
      *(int4*)&a2.h[8] = *(const int4*)&hrow[16 + 8 * hi];   // K=16+8*hi..+7

      // ---- layer 2 -----------------------------------------------------
      v8f d0 = __builtin_amdgcn_wmma_f32_16x16x32_f16(
          false, a2.v, false, b2f[0], (short)0, czero, false, false);
      v8f d1 = __builtin_amdgcn_wmma_f32_16x16x32_f16(
          false, a2.v, false, b2f[1], (short)0, czero, false, false);
      const float bib0 = b2[e * HH + mn];
      const float bib1 = b2[e * HH + mn + 16];
#pragma unroll
      for (int r = 0; r < 8; ++r) {
        h_buf[wib][r + 8 * hi][mn]      = (_Float16)ntanh(d0[r] + bib0);
        h_buf[wib][r + 8 * hi][mn + 16] = (_Float16)ntanh(d1[r] + bib1);
      }
      asm volatile("s_wait_dscnt 0" ::: "memory");

      // ---- layer 3: 32->1 dot + sigmoid (lanes 0..15, one cell each) ---
      if (lane < 16) {
        float acc = b3[e];
#pragma unroll
        for (int k = 0; k < HH; ++k)
          acc += (float)h_buf[wib][mn][k] * W3[e * HH + k];
        xi_buf[wib][t * 16 + mn][e] = nsigmoid(acc);
      }
      asm volatile("s_wait_dscnt 0" ::: "memory");
    }
  }

  // ---------------- Phase 2: per-cell Riemann solve (fp32 VALU) ----------
  const int c = g * 32 + lane;
  if (c >= nc) return;

  const float rhoL = P[c * 6 + 0], rhoR = P[c * 6 + 1];
  const float pL = P[c * 6 + 2], pR = P[c * 6 + 3];
  const float vL = P[c * 6 + 4], vR = P[c * 6 + 5];

  const float pmin = fminf(pL, pR);
  const float csaL = csnd(rhoL, pL);
  const float csaR = csnd(rhoR, pR);

  // expert sweep + argmin |vstarL - vstarR|
  float bestd = 3.402823466e38f;
  float pressCs = 0.f, rhoCL = 0.f, hCL = 0.f, csCL = 0.f, vstarL = 0.f;
  float rhoCR = 0.f, hCR = 0.f, csCR = 0.f, vstarR = 0.f;
  for (int e = 0; e < E_EXP; ++e) {
    const float press = xi_buf[wib][lane][e] * pmin;
    float rl = rhoL * fpow(press * nrcp(pL), 1.f / GAMMA_F);
    float hl = 1.f + GAMMA_F / (GAMMA_F - 1.f) * press * nrcp(rl);
    float cl = nsqrt(GAMMA_F * press * nrcp(rl * hl));
    float vsl = vstar_from_cs(cl, csaL, vL, -1.f);
    float rr = rhoR * fpow(press * nrcp(pR), 1.f / GAMMA_F);
    float hr = 1.f + GAMMA_F / (GAMMA_F - 1.f) * press * nrcp(rr);
    float cr = nsqrt(GAMMA_F * press * nrcp(rr * hr));
    float vsr = vstar_from_cs(cr, csaR, vR, +1.f);
    float d = fabsf(vsl - vsr);
    if (d < bestd) {
      bestd = d; pressCs = press;
      rhoCL = rl; hCL = hl; csCL = cl; vstarL = vsl;
      rhoCR = rr; hCR = hr; csCR = cr; vstarR = vsr;
    }
  }

  const float lambdaC  = 0.5f * (vstarR + vstarL);
  const float lambdaRL = (lambdaC - csCL) * nrcp(1.f - lambdaC * csCL);
  const float lambdaL  = (vL - csaL) * nrcp(1.f - vL * csaL);
  const float lambdaRR = (lambdaC + csCR) * nrcp(1.f + lambdaC * csCR);
  const float lambdaR  = (vR + csaR) * nrcp(1.f + vR * csaR);
  const float WC = nrsq(1.f - lambdaC * lambdaC);
  const float densCL = WC * rhoCL, densCR = WC * rhoCR;
  const float momCL = WC * WC * rhoCL * hCL * lambdaC;
  const float momCR = WC * WC * rhoCR * hCR * lambdaC;
  const float FCL0 = densCL * lambdaC;
  const float FCL1 = densCL * (WC * hCL - 1.f) * lambdaC;
  const float FCL2 = momCL * lambdaC + pressCs;
  const float FCR0 = densCR * lambdaC;
  const float FCR1 = densCR * (WC * hCR - 1.f) * lambdaC;
  const float FCR2 = momCR * lambdaC + pressCs;

  float rho_RL, p_RL, h_RL, v_RL, rho_RR, p_RR, h_RR, v_RR;
  raref_solver(rhoL, pL, vL, +1.f, rho_RL, p_RL, h_RL, v_RL);
  raref_solver(rhoR, pR, vR, -1.f, rho_RR, p_RR, h_RR, v_RR);
  const float WRL = nrsq(1.f - v_RL * v_RL);
  const float WRR = nrsq(1.f - v_RR * v_RR);
  const float densRL = WRL * rho_RL, densRR = WRR * rho_RR;
  const float momRL = WRL * WRL * rho_RL * h_RL * v_RL;
  const float momRR = WRR * WRR * rho_RR * h_RR * v_RR;
  const float FRL0 = densRL * v_RL;
  const float FRL1 = densRL * (WRL * h_RL - 1.f) * v_RL;
  const float FRL2 = momRL * v_RL + p_RL;
  const float FRR0 = densRR * v_RR;
  const float FRR1 = densRR * (WRR * h_RR - 1.f) * v_RR;
  const float FRR2 = momRR * v_RR + p_RR;

  // flux selection chain (same order as reference -> later masks override)
  float fx0 = 0.f, fx1 = 0.f, fx2 = 0.f;
  if (lambdaL >= 0.f) { fx0 = F[c*6+0]; fx1 = F[c*6+2]; fx2 = F[c*6+4]; }
  if (lambdaL < 0.f && lambdaRL >= 0.f) { fx0 = FRL0; fx1 = FRL1; fx2 = FRL2; }
  if (lambdaRL < 0.f && lambdaC > 0.f)  { fx0 = FCL0; fx1 = FCL1; fx2 = FCL2; }
  if (lambdaC <= 0.f && lambdaRR > 0.f) { fx0 = FCR0; fx1 = FCR1; fx2 = FCR2; }
  if (lambdaRR <= 0.f && lambdaR > 0.f) { fx0 = FRR0; fx1 = FRR1; fx2 = FRR2; }
  if (lambdaR <= 0.f) { fx0 = F[c*6+1]; fx1 = F[c*6+3]; fx2 = F[c*6+5]; }

  out[c * 3 + 0] = fx0;
  out[c * 3 + 1] = fx1;
  out[c * 3 + 2] = fx2;
}

extern "C" void kernel_launch(void* const* d_in, const int* in_sizes, int n_in,
                              void* d_out, int out_size, void* d_ws,
                              size_t ws_size, hipStream_t stream) {
  // setup_inputs order: P, U, F, cmax, cmin, W1, b1, W2, b2, W3, b3
  const float* P  = (const float*)d_in[0];
  const float* F  = (const float*)d_in[2];
  const float* W1 = (const float*)d_in[5];
  const float* b1 = (const float*)d_in[6];
  const float* W2 = (const float*)d_in[7];
  const float* b2 = (const float*)d_in[8];
  const float* W3 = (const float*)d_in[9];
  const float* b3 = (const float*)d_in[10];
  const int nc = in_sizes[0] / 6;           // P is (NC,3,2)
  const int ngroups = (nc + 31) / 32;       // 32 cells per wave
  const int blocks = (ngroups + 7) / 8;     // 8 waves per 256-thread block
  fused_riemann_mlp<<<blocks, 256, 0, stream>>>(P, F, W1, b1, W2, b2, W3, b3,
                                                (float*)d_out, nc, ngroups);
}
